// Attention_36636071035300
// MI455X (gfx1250) — matrix operand order repaired and checked by reading, not compiled, not run on hardware
//
#include <hip/hip_runtime.h>
#include <hip/hip_bf16.h>

// ---------------------------------------------------------------------------
// Bahdanau attention fused for MI455X (gfx1250, wave32, WMMA bf16).
// features:[128,196,2048] f32, hidden:[128,512] f32
// out = concat(alpha[128,196], context[128,2048]) f32
// ---------------------------------------------------------------------------

typedef __attribute__((ext_vector_type(16))) __bf16 v16bf;
typedef __attribute__((ext_vector_type(8)))  float  v8f;

#define BB   128
#define LL   196
#define EE   2048
#define DD   512
#define AA   512
#define MTOT (BB * LL)          // 25088
#define NCHUNK (EE / 32)        // 64 K-chunks

__device__ __forceinline__ unsigned short f2bf(float x) {
    union { float f; unsigned int u; } v; v.f = x;
    unsigned int r = v.u + 0x7FFFu + ((v.u >> 16) & 1u);   // round-to-nearest-even
    return (unsigned short)(r >> 16);
}

__device__ __forceinline__ float fast_tanh(float x) {
#if __has_builtin(__builtin_amdgcn_tanhf)
    return __builtin_amdgcn_tanhf(x);        // V_TANH_F32 trans op
#else
    return tanhf(x);
#endif
}

__device__ __forceinline__ v8f wmma_bf16(v16bf a, v16bf b, v8f c) {
    return __builtin_amdgcn_wmma_f32_16x16x32_bf16(false, a, false, b,
                                                   (short)0, c, false, false);
}

// ---------------------------------------------------------------------------
// Stage 0: pre-swizzle W2_w (f32 [2048,512]) -> bf16 fragment-layout buffer.
// Layout: [chunk=k/32][t=n/16][lane][16 u16], lane = (n&15) + 16*(kk>=16),
// within lane: u16 index = ((kk&15)>>1)*2 + (kk&1)  -> a wave's B fragment is
// 1024 contiguous bytes, 32 B per lane (2 x global_load_b128, coalesced).
// grid 1024, block 256 (4 elements per thread)
// ---------------------------------------------------------------------------
__global__ __launch_bounds__(256) void w2_swizzle(
    const float* __restrict__ W2_w, unsigned short* __restrict__ W2bf)
{
    const int idx4 = blockIdx.x * 256 + threadIdx.x;    // float4 slot, 262144 total
    float4 f = ((const float4*)W2_w)[idx4];
    const int k    = idx4 >> 7;                         // 128 float4 per row of 512
    const int nq   = (idx4 & 127) * 4;
    const int ch   = k >> 5;
    const int kk   = k & 31;
    const int bl16 = (kk >= 16) ? 16 : 0;
    const int j2   = ((kk & 15) >> 1) * 2 + (kk & 1);
    const float fv[4] = { f.x, f.y, f.z, f.w };
    #pragma unroll
    for (int q = 0; q < 4; ++q) {
        const int n = nq + q;
        W2bf[((size_t)ch * 32 + (n >> 4)) * 512 + ((n & 15) + bl16) * 16 + j2] = f2bf(fv[q]);
    }
}

// ---------------------------------------------------------------------------
// Stage A: w1_ah[b][n] = sum_k hidden[b][k] * W1_w[k][n] + W1_b[n]
// grid (128, 2), block 256
// ---------------------------------------------------------------------------
__global__ __launch_bounds__(256) void w1_kernel(
    const float* __restrict__ hidden, const float* __restrict__ W1_w,
    const float* __restrict__ W1_b, float* __restrict__ w1_ah)
{
    __shared__ float h_sh[DD];
    const int b = blockIdx.x;
    const int n = blockIdx.y * 256 + threadIdx.x;
    h_sh[threadIdx.x]       = hidden[b * DD + threadIdx.x];
    h_sh[threadIdx.x + 256] = hidden[b * DD + threadIdx.x + 256];
    __syncthreads();
    float acc = W1_b[n];
    #pragma unroll 8
    for (int k = 0; k < DD; ++k)
        acc = fmaf(h_sh[k], W1_w[k * AA + n], acc);
    w1_ah[b * AA + n] = acc;
}

// ---------------------------------------------------------------------------
// Stage B: fused scores[m] = A_b + sum_n tanh((features@W2_w)[m][n]
//                                             + w1_ah[b(m)][n] + W2_b[n]) * A_w[n]
// Workgroup = 64 M-rows x all 512 N. 8 waves; wave owns a 64-wide N strip
// (4x4 grid of 16x16 WMMA tiles). K chunked by 32.
//  - A (features) staged through double-buffered LDS (f32->bf16, pre-swizzled
//    to fragment layout; 1 barrier per chunk, next chunk prefetched into regs)
//  - B fragments loaded straight from the pre-swizzled bf16 W2 buffer in L2.
// grid 392, block 256
// ---------------------------------------------------------------------------
__global__ __launch_bounds__(256) void score_gemm(
    const float* __restrict__ features,        // [MTOT, EE]
    const unsigned short* __restrict__ W2bf,   // pre-swizzled bf16
    const float* __restrict__ W2_b,            // [AA]
    const float* __restrict__ w1_ah,           // [BB, AA]
    const float* __restrict__ A_w,             // [AA]
    const float* __restrict__ A_b,             // scalar
    float* __restrict__ scores)                // [MTOT]
{
    __shared__ __align__(32) unsigned short ldsA[2][4 * 512];   // 2 x 4 KB
    __shared__ float ldsS[64];

    const int tid  = threadIdx.x;
    const int wave = tid >> 5;          // 0..7 -> N strip wave*64
    const int lane = tid & 31;
    const int m0   = blockIdx.x * 64;

    if (tid < 64) ldsS[tid] = 0.0f;

    // Two A-staging slots per thread (64 rows x 8 float4-slots = 512 slots).
    const int row0 = tid >> 3,           kq0 = (tid & 7) * 4;
    const int row1 = (tid + 256) >> 3,   kq1 = (tid & 7) * 4;   // same kq pattern
    const int base0 = (row0 >> 4) * 512 + ((row0 & 15) + ((kq0 & 16) ? 16 : 0)) * 16
                    + (kq0 & 15);
    const int base1 = (row1 >> 4) * 512 + ((row1 & 15) + ((kq1 & 16) ? 16 : 0)) * 16
                    + (kq1 & 15);
    const float4* gA0 = (const float4*)(features + (size_t)(m0 + row0) * EE + kq0);
    const float4* gA1 = (const float4*)(features + (size_t)(m0 + row1) * EE + kq1);

    float4 fA0 = gA0[0];                // chunk 0 prefetch
    float4 fA1 = gA1[0];

    v8f c[4][4] = {};                   // fp32 accumulators, 16 tiles per wave

    for (int ch = 0; ch < NCHUNK; ++ch) {
        unsigned short* bufA = ldsA[ch & 1];
        ushort4 h0, h1;
        h0.x = f2bf(fA0.x); h0.y = f2bf(fA0.y); h0.z = f2bf(fA0.z); h0.w = f2bf(fA0.w);
        h1.x = f2bf(fA1.x); h1.y = f2bf(fA1.y); h1.z = f2bf(fA1.z); h1.w = f2bf(fA1.w);
        *(ushort4*)(bufA + base0) = h0;
        *(ushort4*)(bufA + base1) = h1;
        __syncthreads();

        if (ch + 1 < NCHUNK) {          // prefetch next A chunk (k += 32 -> +8 float4)
            fA0 = gA0[(ch + 1) * 8];
            fA1 = gA1[(ch + 1) * 8];
        }

        // B fragments: contiguous 32 B/lane from pre-swizzled L2-resident buffer
        const unsigned short* bp =
            W2bf + ((size_t)ch * 32 + wave * 4) * 512 + lane * 16;
        v16bf bf0 = *(const v16bf*)(bp);
        v16bf bf1 = *(const v16bf*)(bp + 512);
        v16bf bf2 = *(const v16bf*)(bp + 1024);
        v16bf bf3 = *(const v16bf*)(bp + 1536);

        #pragma unroll
        for (int s = 0; s < 4; ++s) {
            v16bf af = *(const v16bf*)(bufA + s * 512 + lane * 16);
            c[s][0] = wmma_bf16(af, bf0, c[s][0]);
            c[s][1] = wmma_bf16(af, bf1, c[s][1]);
            c[s][2] = wmma_bf16(af, bf2, c[s][2]);
            c[s][3] = wmma_bf16(af, bf3, c[s][3]);
        }
    }

    // ---- epilogue: bias + tanh + A_w dot, reduce over N ----
    // C layout: VGPR r, lanes 0-15: (M=r, N=lane); lanes 16-31: (M=8+r, N=lane-16)
    const int nlo = lane & 15;
    const int hi  = lane >> 4;
    #pragma unroll
    for (int s = 0; s < 4; ++s) {
        float part[8];
        #pragma unroll
        for (int r = 0; r < 8; ++r) {
            const int m_g = m0 + s * 16 + r + hi * 8;
            const int b   = m_g / LL;
            float acc = 0.0f;
            #pragma unroll
            for (int t = 0; t < 4; ++t) {
                const int n = wave * 64 + t * 16 + nlo;
                const float bias = w1_ah[b * AA + n] + W2_b[n];
                acc = fmaf(fast_tanh(c[s][t][r] + bias), A_w[n], acc);
            }
            part[r] = acc;
        }
        #pragma unroll
        for (int mask = 1; mask < 16; mask <<= 1) {
            #pragma unroll
            for (int r = 0; r < 8; ++r)
                part[r] += __shfl_xor(part[r], mask, 32);
        }
        if (nlo == 0) {
            #pragma unroll
            for (int r = 0; r < 8; ++r)
                atomicAdd(&ldsS[s * 16 + r + hi * 8], part[r]);
        }
    }
    __syncthreads();
    if (tid < 64) scores[m0 + tid] = ldsS[tid] + A_b[0];
}

// ---------------------------------------------------------------------------
// Stage C: softmax over L per batch row.  grid 128, block 256
// ---------------------------------------------------------------------------
__global__ __launch_bounds__(256) void softmax_kernel(
    const float* __restrict__ scores, float* __restrict__ alpha)
{
    __shared__ float red[256];
    const int b = blockIdx.x, tid = threadIdx.x;
    const float v = (tid < LL) ? scores[b * LL + tid] : -3.4e38f;
    red[tid] = v; __syncthreads();
    for (int s = 128; s > 0; s >>= 1) {
        if (tid < s) red[tid] = fmaxf(red[tid], red[tid + s]);
        __syncthreads();
    }
    const float mx = red[0]; __syncthreads();
    const float e = (tid < LL) ? __expf(v - mx) : 0.0f;
    red[tid] = e; __syncthreads();
    for (int s = 128; s > 0; s >>= 1) {
        if (tid < s) red[tid] += red[tid + s];
        __syncthreads();
    }
    const float inv = 1.0f / red[0];
    if (tid < LL) alpha[b * LL + tid] = e * inv;
}

// ---------------------------------------------------------------------------
// Stage D: context[b][e] = sum_l alpha[b][l] * features[b][l][e]
// grid (128, 8), block 256
// ---------------------------------------------------------------------------
__global__ __launch_bounds__(256) void context_kernel(
    const float* __restrict__ features, const float* __restrict__ alpha,
    float* __restrict__ ctx)
{
    __shared__ float a_sh[LL];
    const int b = blockIdx.x;
    const int e = blockIdx.y * 256 + threadIdx.x;
    if (threadIdx.x < LL) a_sh[threadIdx.x] = alpha[b * LL + threadIdx.x];
    __syncthreads();
    float acc = 0.0f;
    const float* f = features + (size_t)b * LL * EE + e;
    #pragma unroll 4
    for (int l = 0; l < LL; ++l)
        acc = fmaf(a_sh[l], f[(size_t)l * EE], acc);
    ctx[b * EE + e] = acc;
}

// ---------------------------------------------------------------------------
extern "C" void kernel_launch(void* const* d_in, const int* in_sizes, int n_in,
                              void* d_out, int out_size, void* d_ws, size_t ws_size,
                              hipStream_t stream) {
    const float* features = (const float*)d_in[0];
    const float* hidden   = (const float*)d_in[1];
    const float* W1_w     = (const float*)d_in[2];
    const float* W1_b     = (const float*)d_in[3];
    const float* W2_w     = (const float*)d_in[4];
    const float* W2_b     = (const float*)d_in[5];
    const float* A_w      = (const float*)d_in[6];
    const float* A_b      = (const float*)d_in[7];

    float* alpha   = (float*)d_out;             // [128*196]
    float* context = (float*)d_out + BB * LL;   // [128*2048]

    float* w1_ah  = (float*)d_ws;                               // 128*512 f32
    float* scores = (float*)d_ws + BB * AA;                     // 25088 f32
    unsigned short* W2bf = (unsigned short*)((float*)d_ws + BB * AA + MTOT); // 1M bf16

    w2_swizzle<<<dim3(1024), 256, 0, stream>>>(W2_w, W2bf);
    w1_kernel<<<dim3(BB, 2), 256, 0, stream>>>(hidden, W1_w, W1_b, w1_ah);
    score_gemm<<<dim3(MTOT / 64), 256, 0, stream>>>(features, W2bf, W2_b,
                                                    w1_ah, A_w, A_b, scores);
    softmax_kernel<<<dim3(BB), 256, 0, stream>>>(scores, alpha);
    context_kernel<<<dim3(BB, 8), 256, 0, stream>>>(features, alpha, context);
}
